// AttnBlock2D_6768868458869
// MI455X (gfx1250) — compile-verified, matrix operand order checked by reading, not run on hardware
//
#include <hip/hip_runtime.h>

#define C_DIM 512
#define N_DIM 4096
#define T_DIM 4
#define GROUPS 32

typedef __attribute__((ext_vector_type(16))) __bf16 v16bf;
typedef __attribute__((ext_vector_type(8)))  __bf16 v8bf;
typedef __attribute__((ext_vector_type(8)))  float  v8f;
typedef __attribute__((ext_vector_type(4)))  int    v4i;

// ---------------------------------------------------------------------------
// ISA 7.12.2 fragment layout (16-bit A, 16x32): lane group g = lane>>4 holds
// K = [8g, 8g+8) in VGPRs 0..3 and K = [16+8g, 16+8g+8) in VGPRs 4..7, i.e.
// TWO CONTIGUOUS 8-ELEMENT RUNS:
//   * A-fragments from row-major storage = two 16B loads.
//   * B-matrices stored pre-permuted ("bperm"): per (32K x 16N) block each
//     lane's 16 B-elements are one contiguous 32B run -> one v16bf load.
// ---------------------------------------------------------------------------
__device__ __forceinline__ v16bf concat8(v8bf lo, v8bf hi) {
    return __builtin_shufflevector(lo, hi, 0, 1, 2, 3, 4, 5, 6, 7,
                                   8, 9, 10, 11, 12, 13, 14, 15);
}

__device__ __forceinline__ v16bf load_afrag(const __bf16* row, int g) {
    v8bf lo = *(const v8bf*)(row + 8 * g);
    v8bf hi = *(const v8bf*)(row + 16 + 8 * g);
    return concat8(lo, hi);
}

__device__ __forceinline__ v16bf load_bfrag(const __bf16* p) {
    return *(const v16bf*)p;
}

// flat element index of (k, n) inside a bperm-laid-out (Kdim x Ndim) matrix
__device__ __forceinline__ size_t bperm_idx(int k, int n, int kdim) {
    int km = k & 31;
    int g  = (km >> 3) & 1;
    int e  = (km & 7) | ((km & 16) ? 8 : 0);
    return ((((size_t)(n >> 4) * (kdim >> 5)) + (k >> 5)) << 9) +
           (size_t)(((n & 15) + (g << 4)) * 16 + e);
}

static __device__ __forceinline__ v8f wmma_bf16(v16bf a, v16bf b, v8f c) {
    return __builtin_amdgcn_wmma_f32_16x16x32_bf16(false, a, false, b, (short)0, c, false, false);
}

// ---------------------------------------------------------------------------
// Kernel 0: convert the four 512x512 fp32 weight matrices to bf16 once.
// ---------------------------------------------------------------------------
__global__ __launch_bounds__(256) void wcast_kernel(const float* __restrict__ wq,
                                                    const float* __restrict__ wk,
                                                    const float* __restrict__ wv,
                                                    const float* __restrict__ wp,
                                                    __bf16* __restrict__ wb) {
    const size_t M = (size_t)C_DIM * C_DIM;
    size_t idx = (size_t)blockIdx.x * 256 + threadIdx.x;
    if (idx >= 4 * M) return;
    int s = (int)(idx / M);
    size_t r = idx % M;
    const float* src = (s == 0) ? wq : (s == 1) ? wk : (s == 2) ? wv : wp;
    wb[idx] = (__bf16)src[r];
}

// ---------------------------------------------------------------------------
// Kernel 1: GroupNorm statistics (group = contiguous 1MB chunk of x).
// ---------------------------------------------------------------------------
__global__ __launch_bounds__(256) void gn_stats_kernel(const float* __restrict__ x,
                                                       float* __restrict__ stats) {
    const int g = blockIdx.x;
    const int M = 16 * T_DIM * N_DIM;
    const float* p = x + (size_t)g * M;
    float s = 0.f, ss = 0.f;
    for (int idx = threadIdx.x; idx < M; idx += 256) {
        float v = p[idx];
        s += v;
        ss += v * v;
    }
    __shared__ float rs[256], rss[256];
    rs[threadIdx.x] = s;
    rss[threadIdx.x] = ss;
    __syncthreads();
    for (int st = 128; st > 0; st >>= 1) {
        if (threadIdx.x < st) {
            rs[threadIdx.x] += rs[threadIdx.x + st];
            rss[threadIdx.x] += rss[threadIdx.x + st];
        }
        __syncthreads();
    }
    if (threadIdx.x == 0) {
        float mu = rs[0] / (float)M;
        float var = rss[0] / (float)M - mu * mu;
        stats[g * 2 + 0] = mu;
        stats[g * 2 + 1] = rsqrtf(var + 1e-6f);
    }
}

// ---------------------------------------------------------------------------
// Kernel 2: GroupNorm apply -> hn in bperm layout per frame (K=c, N=n).
// ---------------------------------------------------------------------------
__global__ __launch_bounds__(256) void gn_apply_kernel(const float* __restrict__ x,
                                                       const float* __restrict__ gamma,
                                                       const float* __restrict__ beta,
                                                       const float* __restrict__ stats,
                                                       __bf16* __restrict__ hn) {
    const int TN = T_DIM * N_DIM;
    size_t idx = (size_t)blockIdx.x * 256 + threadIdx.x;
    if (idx >= (size_t)C_DIM * TN) return;
    int c = (int)(idx / TN);
    int r = (int)(idx % TN);
    int t = r / N_DIM;
    int n = r % N_DIM;
    int g = c >> 4;
    float mu = stats[2 * g], rsg = stats[2 * g + 1];
    float v = (x[idx] - mu) * rsg * gamma[c] + beta[c];
    hn[(size_t)t * C_DIM * N_DIM + bperm_idx(c, n, C_DIM)] = (__bf16)v;
}

// ---------------------------------------------------------------------------
// Kernel 3: QKV projections. blockIdx.z = t*3 + s. Each wave computes one
// 16-row o-tile x FOUR 16-col i-tiles (A-fragment reused 4x per K-step).
// Outputs: q row-major (t,i,c) * 1/sqrt(C)  (A-matrix of S-GEMM)
//          k bperm (K=c, N=j)               (B-matrix of S-GEMM)
//          v bperm (K=j, N=c)               (B-matrix of PV-GEMM)
// ---------------------------------------------------------------------------
__global__ __launch_bounds__(128) void qkv_gemm_kernel(
    const __bf16* __restrict__ wb,
    const float* __restrict__ bq, const float* __restrict__ bk, const float* __restrict__ bv,
    const __bf16* __restrict__ hn,
    __bf16* __restrict__ qbuf, __bf16* __restrict__ kbuf, __bf16* __restrict__ vbuf) {
    const int z = blockIdx.z;
    const int t = z / 3, s = z - 3 * t;
    const __bf16* W   = wb + (size_t)s * C_DIM * C_DIM;
    const float* bias = (s == 0) ? bq : ((s == 1) ? bk : bv);

    const int lane = threadIdx.x & 31;
    const int wave = threadIdx.x >> 5;
    const int m16 = lane & 15, g = lane >> 4;
    const int o0 = (blockIdx.y * 4 + wave) * 16;
    const int nt0 = blockIdx.x * 4;                 // first of 4 i-tiles
    const __bf16* hf = hn + (size_t)t * C_DIM * N_DIM;
    const __bf16* Wrow = W + (size_t)(o0 + m16) * C_DIM;
    const __bf16* hblk = hf + (((size_t)nt0 * (C_DIM >> 5)) << 9) + (size_t)lane * 16;
    const size_t NTSTRIDE = (size_t)(C_DIM >> 5) << 9;   // elems per nt row of blocks

    v8f zero = {};
    v8f acc[4] = {zero, zero, zero, zero};
    for (int kc = 0; kc < C_DIM; kc += 32) {
        v16bf a = load_afrag(Wrow + kc, g);
        const __bf16* bb = hblk + ((size_t)(kc >> 5) << 9);
#pragma unroll
        for (int ct = 0; ct < 4; ++ct)
            acc[ct] = wmma_bf16(a, load_bfrag(bb + (size_t)ct * NTSTRIDE), acc[ct]);
    }

    const float scale = rsqrtf((float)C_DIM);
    const size_t frame = (size_t)t * C_DIM * N_DIM;
    float bv8[8];
#pragma unroll
    for (int r = 0; r < 8; ++r) bv8[r] = bias[o0 + r + 8 * g];

#pragma unroll
    for (int ct = 0; ct < 4; ++ct) {
        const int i = (nt0 + ct) * 16 + m16;
        if (s == 0) {
            v8bf qv;
#pragma unroll
            for (int r = 0; r < 8; ++r) qv[r] = (__bf16)((acc[ct][r] + bv8[r]) * scale);
            *(v8bf*)(qbuf + frame + (size_t)i * C_DIM + o0 + 8 * g) = qv;
        } else if (s == 1) {
            v8bf kv;
#pragma unroll
            for (int r = 0; r < 8; ++r) kv[r] = (__bf16)(acc[ct][r] + bv8[r]);
            __bf16* dst = kbuf + frame +
                          ((((size_t)(nt0 + ct) * (C_DIM >> 5)) + (o0 >> 5)) << 9) +
                          (size_t)lane * 16 + ((o0 & 16) ? 8 : 0);
            *(v8bf*)dst = kv;
        } else {
            // v: K-dim = i (fixed per lane), N-dim = o (varies with r) -> scatter
            int km_i = i & 31;
            int g_i = (km_i >> 3) & 1;
            int e_i = (km_i & 7) | ((km_i & 16) ? 8 : 0);
            size_t vbase = frame + ((((size_t)(o0 >> 4) * (N_DIM >> 5)) + (i >> 5)) << 9) + e_i;
#pragma unroll
            for (int r = 0; r < 8; ++r)
                vbuf[vbase + (size_t)((r + 8 * g) + 16 * g_i) * 16] =
                    (__bf16)(acc[ct][r] + bv8[r]);
        }
    }
}

// ---------------------------------------------------------------------------
// Kernel 4: fused flash attention. 8 waves, 32 query rows/block.
// Q staged in LDS via async-load-to-LDS (ASYNCcnt path) when available.
// ---------------------------------------------------------------------------
__global__ __launch_bounds__(256) void flash_attn_kernel(const __bf16* __restrict__ qbuf,
                                                         const __bf16* __restrict__ kbuf,
                                                         const __bf16* __restrict__ vbuf,
                                                         __bf16* __restrict__ obuf) {
    __shared__ alignas(16) __bf16 sQ[32][C_DIM];
    __shared__ alignas(16) float  sS[32][132];
    __shared__ alignas(16) __bf16 sP[32][128];
    __shared__ float sM[32], sL[32], sAl[32];

    const int t = blockIdx.y;
    const int i0 = blockIdx.x * 32;
    const __bf16* qf = qbuf + (size_t)t * N_DIM * C_DIM;
    const __bf16* kf = kbuf + (size_t)t * C_DIM * N_DIM;
    const __bf16* vf = vbuf + (size_t)t * N_DIM * C_DIM;

#if defined(__has_builtin) && __has_builtin(__builtin_amdgcn_global_load_async_to_lds_b128)
#define AS1 __attribute__((address_space(1)))
#define AS3 __attribute__((address_space(3)))
    for (int idx = threadIdx.x; idx < 2048; idx += 256) {
        int r = idx >> 6, c8 = (idx & 63) << 3;
        const __bf16* gp = qf + (size_t)(i0 + r) * C_DIM + c8;
        __builtin_amdgcn_global_load_async_to_lds_b128(
            (AS1 v4i*)(uintptr_t)gp, (AS3 v4i*)&sQ[r][c8], 0, 0);
    }
#if __has_builtin(__builtin_amdgcn_s_wait_asynccnt)
    __builtin_amdgcn_s_wait_asynccnt(0);
#else
    asm volatile("s_wait_asynccnt 0x0" ::: "memory");
#endif
#else
    for (int idx = threadIdx.x; idx < 2048; idx += 256) {
        int r = idx >> 6, c8 = (idx & 63) << 3;
        *(v8bf*)&sQ[r][c8] = *(const v8bf*)(qf + (size_t)(i0 + r) * C_DIM + c8);
    }
#endif
    if (threadIdx.x < 32) { sM[threadIdx.x] = -1e30f; sL[threadIdx.x] = 0.f; }
    __syncthreads();

    const int wave = threadIdx.x >> 5;
    const int lane = threadIdx.x & 31;
    const int m16 = lane & 15, g = lane >> 4;
    const int c0 = wave * 64;
    const size_t CTSTRIDE = (size_t)(N_DIM >> 5) << 9;   // bperm nt-row stride (elems)

    v8f zero = {};
    v8f acc[2][4];
#pragma unroll
    for (int a0 = 0; a0 < 2; ++a0)
#pragma unroll
        for (int a1 = 0; a1 < 4; ++a1) acc[a0][a1] = zero;

    for (int j0 = 0; j0 < N_DIM; j0 += 128) {
        // --- S = Q x K : wave w computes j-columns [j0+16w, +16); B-fragment
        // (K-matrix) hoisted across the two M tiles.
        const int jc = j0 + wave * 16;
        const __bf16* kblk = kf + (((size_t)(jc >> 4) * (C_DIM >> 5)) << 9) + (size_t)lane * 16;
        v8f s0 = zero, s1 = zero;
        for (int kc = 0; kc < C_DIM; kc += 32) {
            v16bf b  = load_bfrag(kblk + ((size_t)(kc >> 5) << 9));
            v16bf a0 = load_afrag(&sQ[m16][kc], g);
            v16bf a1 = load_afrag(&sQ[16 + m16][kc], g);
            s0 = wmma_bf16(a0, b, s0);
            s1 = wmma_bf16(a1, b, s1);
        }
#pragma unroll
        for (int r = 0; r < 8; ++r) {
            sS[r + 8 * g][wave * 16 + m16]      = s0[r];
            sS[16 + r + 8 * g][wave * 16 + m16] = s1[r];
        }
        if (j0 + 128 < N_DIM)
            __builtin_prefetch(kf + ((((size_t)(jc + 128) >> 4) * (C_DIM >> 5)) << 9) +
                                   (size_t)lane * 16, 0, 0);
        __syncthreads();

        // --- online softmax: one thread per query row
        if (threadIdx.x < 32) {
            const int r = threadIdx.x;
            float mo = sM[r], mn = mo;
            for (int j = 0; j < 128; ++j) mn = fmaxf(mn, sS[r][j]);
            float al = __expf(mo - mn);
            float ls = 0.f;
            for (int j = 0; j < 128; ++j) {
                float p = __expf(sS[r][j] - mn);
                sP[r][j] = (__bf16)p;
                ls += p;
            }
            sM[r] = mn;
            sL[r] = sL[r] * al + ls;
            sAl[r] = al;
        }
        __syncthreads();

        // --- rescale O by alpha
#pragma unroll
        for (int mt = 0; mt < 2; ++mt) {
#pragma unroll
            for (int r = 0; r < 8; ++r) {
                float al = sAl[mt * 16 + r + 8 * g];
#pragma unroll
                for (int ct = 0; ct < 4; ++ct) acc[mt][ct][r] *= al;
            }
        }
        // --- O += P x V : load 4 V-fragments once per kj, reuse for both
        // M tiles; A (P) fragment reused across 4 c-tiles. 12 loads / 8 WMMA.
        const __bf16* vblk = vf +
            ((((size_t)(c0 >> 4) * (N_DIM >> 5)) + (j0 >> 5)) << 9) + (size_t)lane * 16;
#pragma unroll
        for (int kj = 0; kj < 128; kj += 32) {
            v16bf bv4[4];
#pragma unroll
            for (int ct = 0; ct < 4; ++ct)
                bv4[ct] = load_bfrag(vblk + (size_t)ct * CTSTRIDE + ((size_t)(kj >> 5) << 9));
#pragma unroll
            for (int mt = 0; mt < 2; ++mt) {
                v16bf a = load_afrag(&sP[mt * 16 + m16][kj], g);
#pragma unroll
                for (int ct = 0; ct < 4; ++ct)
                    acc[mt][ct] = wmma_bf16(a, bv4[ct], acc[mt][ct]);
            }
        }
        __syncthreads();
    }

    // --- finalize: O /= l, store to obuf in bperm layout (K=c, N=i)
    const size_t frame = (size_t)t * C_DIM * N_DIM;
#pragma unroll
    for (int mt = 0; mt < 2; ++mt) {
        const int nt = (i0 + mt * 16) >> 4;
#pragma unroll
        for (int ct = 0; ct < 4; ++ct) {
            int c = c0 + ct * 16 + m16;
            int km_c = c & 31;
            int g_c = (km_c >> 3) & 1;
            int e_c = (km_c & 7) | ((km_c & 16) ? 8 : 0);
            size_t base = frame + (((size_t)nt * (C_DIM >> 5) + (c >> 5)) << 9) + e_c;
#pragma unroll
            for (int r = 0; r < 8; ++r) {
                float inv = 1.0f / sL[mt * 16 + r + 8 * g];
                obuf[base + (size_t)((r + 8 * g) + 16 * g_c) * 16] =
                    (__bf16)(acc[mt][ct][r] * inv);
            }
        }
    }
}

// ---------------------------------------------------------------------------
// Kernel 5: output projection + bias + residual -> fp32 (C,T,H,W).
// Wave computes 1 o-tile x 4 i-tiles (A reused 4x per K-step).
// ---------------------------------------------------------------------------
__global__ __launch_bounds__(128) void proj_kernel(const __bf16* __restrict__ wb,
                                                   const float* __restrict__ bp,
                                                   const __bf16* __restrict__ obuf,
                                                   const float* __restrict__ x,
                                                   float* __restrict__ out) {
    const int t = blockIdx.z;
    const int lane = threadIdx.x & 31;
    const int wave = threadIdx.x >> 5;
    const int m16 = lane & 15, g = lane >> 4;
    const int o0 = (blockIdx.y * 4 + wave) * 16;
    const int nt0 = blockIdx.x * 4;
    const __bf16* Wrow = wb + (size_t)(o0 + m16) * C_DIM;
    const __bf16* oblk = obuf + (size_t)t * C_DIM * N_DIM +
                         (((size_t)nt0 * (C_DIM >> 5)) << 9) + (size_t)lane * 16;
    const size_t NTSTRIDE = (size_t)(C_DIM >> 5) << 9;

    v8f zero = {};
    v8f acc[4] = {zero, zero, zero, zero};
    for (int kc = 0; kc < C_DIM; kc += 32) {
        v16bf a = load_afrag(Wrow + kc, g);
        const __bf16* bb = oblk + ((size_t)(kc >> 5) << 9);
#pragma unroll
        for (int ct = 0; ct < 4; ++ct)
            acc[ct] = wmma_bf16(a, load_bfrag(bb + (size_t)ct * NTSTRIDE), acc[ct]);
    }

#pragma unroll
    for (int ct = 0; ct < 4; ++ct) {
#pragma unroll
        for (int r = 0; r < 8; ++r) {
            int o = o0 + r + 8 * g;
            int i = (nt0 + ct) * 16 + m16;
            size_t oi = ((size_t)o * T_DIM + t) * N_DIM + i;
            out[oi] = x[oi] + acc[ct][r] + bp[o];
        }
    }
}

// ---------------------------------------------------------------------------
extern "C" void kernel_launch(void* const* d_in, const int* in_sizes, int n_in,
                              void* d_out, int out_size, void* d_ws, size_t ws_size,
                              hipStream_t stream) {
    const float* x     = (const float*)d_in[0];
    const float* gamma = (const float*)d_in[1];
    const float* beta  = (const float*)d_in[2];
    const float* wq    = (const float*)d_in[3];
    const float* bq    = (const float*)d_in[4];
    const float* wk    = (const float*)d_in[5];
    const float* bk    = (const float*)d_in[6];
    const float* wv    = (const float*)d_in[7];
    const float* bv    = (const float*)d_in[8];
    const float* wp    = (const float*)d_in[9];
    const float* bp    = (const float*)d_in[10];
    float* out = (float*)d_out;

    char* ws = (char*)d_ws;
    float* stats = (float*)ws;                                   // 64 floats
    const size_t WBSZ = (size_t)4 * C_DIM * C_DIM * sizeof(__bf16);          // 2 MB
    const size_t MAT  = (size_t)T_DIM * C_DIM * N_DIM * sizeof(__bf16);      // 16 MB
    size_t off = 1024;
    __bf16* wbuf = (__bf16*)(ws + off); off += WBSZ;
    __bf16* hn = (__bf16*)(ws + off); off += MAT;
    __bf16* qb = (__bf16*)(ws + off); off += MAT;
    __bf16* kb = (__bf16*)(ws + off); off += MAT;
    __bf16* vb = (__bf16*)(ws + off); off += MAT;
    __bf16* ob = (__bf16*)(ws + off); off += MAT;

    wcast_kernel<<<(4 * C_DIM * C_DIM) / 256, 256, 0, stream>>>(wq, wk, wv, wp, wbuf);
    gn_stats_kernel<<<GROUPS, 256, 0, stream>>>(x, stats);
    gn_apply_kernel<<<(C_DIM * T_DIM * N_DIM) / 256, 256, 0, stream>>>(x, gamma, beta, stats, hn);
    qkv_gemm_kernel<<<dim3(N_DIM / 64, C_DIM / 64, 3 * T_DIM), 128, 0, stream>>>(
        wbuf, bq, bk, bv, hn, qb, kb, vb);
    flash_attn_kernel<<<dim3(N_DIM / 32, T_DIM), 256, 0, stream>>>(qb, kb, vb, ob);
    proj_kernel<<<dim3(N_DIM / 64, C_DIM / 64, T_DIM), 128, 0, stream>>>(wbuf + 3 * C_DIM * C_DIM, bp, ob, x, out);
}
